// GeneratorBlock_83219286327501
// MI455X (gfx1250) — compile-verified, matrix operand order checked by reading, not run on hardware
//
#include <hip/hip_runtime.h>
#include <hip/hip_bf16.h>

typedef __attribute__((ext_vector_type(16))) _Float16 v16h;
typedef __attribute__((ext_vector_type(8)))  float    v8f;
typedef __attribute__((__vector_size__(4 * sizeof(int)))) int v4i;

#define GLOBAL_AS __attribute__((address_space(1)))
#define LDS_AS    __attribute__((address_space(3)))

#define LRELU_SLOPE 0.2f

#ifndef __has_builtin
#define __has_builtin(x) 0
#endif
#if __has_builtin(__builtin_amdgcn_global_load_async_to_lds_b128) && \
    __has_builtin(__builtin_amdgcn_s_wait_asynccnt)
#define USE_ASYNC 1
#else
#define USE_ASYNC 0
#endif

// ---------------------------------------------------------------------------
// style = istyle @ W.T + b ; store (style + 1)
// grid: (8), block: (256)   L = 512
// ---------------------------------------------------------------------------
__global__ void style_kernel(const float* __restrict__ istyle,
                             const float* __restrict__ w,
                             const float* __restrict__ bb,
                             float* __restrict__ sp, int L) {
  int b = blockIdx.x, c = threadIdx.x;
  const float* xi = istyle + (size_t)b * L;
  const float* wr = w + (size_t)c * L;
  float s = bb[c];
  for (int l = 0; l < L; ++l) s += xi[l] * wr[l];
  sp[b * 256 + c] = s + 1.0f;
}

// ---------------------------------------------------------------------------
// wsq[f,c] = sum over 3x3 of w^2    (65536 threads)
// ---------------------------------------------------------------------------
__global__ void wsq_kernel(const float* __restrict__ w, float* __restrict__ wsq) {
  int idx = blockIdx.x * blockDim.x + threadIdx.x;
  const float* p = w + (size_t)idx * 9;
  float s = 0.f;
#pragma unroll
  for (int r = 0; r < 9; ++r) { float v = p[r]; s += v * v; }
  wsq[idx] = s;
}

// ---------------------------------------------------------------------------
// d[b,f] = rsqrt( sum_c sp[b,c]^2 * wsq[f,c] + eps )
// grid: (8), block: (256)
// ---------------------------------------------------------------------------
__global__ void dcoef_kernel(const float* __restrict__ sp,
                             const float* __restrict__ wsq,
                             float* __restrict__ d) {
  int b = blockIdx.x, f = threadIdx.x;
  const float* wr = wsq + (size_t)f * 256;
  const float* sr = sp + (size_t)b * 256;
  float acc = 0.f;
  for (int c = 0; c < 256; ++c) { float s = sr[c]; acc += s * s * wr[c]; }
  d[b * 256 + f] = rsqrtf(acc + 1e-8f);
}

// ---------------------------------------------------------------------------
// wmod[b,f,k] = f16( w[f,c,r] * sp[b,c] * d[b,f] ), k = c*9 + r, row stride Kst
// total threads = 8*256*2304
// ---------------------------------------------------------------------------
__global__ void modw_kernel(const float* __restrict__ w,
                            const float* __restrict__ sp,
                            const float* __restrict__ d,
                            _Float16* __restrict__ out, int Kst) {
  size_t idx = (size_t)blockIdx.x * blockDim.x + threadIdx.x;
  int k = (int)(idx % 2304);
  size_t t = idx / 2304;
  int f = (int)(t & 255);
  int b = (int)(t >> 8);
  int c = k / 9, r = k - c * 9;
  float v = w[((size_t)f * 256 + c) * 9 + r] * sp[b * 256 + c] * d[b * 256 + f];
  out[((size_t)b * 256 + f) * Kst + k] = (_Float16)v;
}

// ---------------------------------------------------------------------------
// append unmodulated 1x1 weights at K-offset 2304 of wmod2 (folded residual)
// total threads = 8*256*256
// ---------------------------------------------------------------------------
__global__ void tail_kernel(const float* __restrict__ w1x1,
                            _Float16* __restrict__ wmod2) {
  int idx = blockIdx.x * blockDim.x + threadIdx.x;
  int c = idx & 255;
  int t = idx >> 8;
  int f = t & 255;
  int b = t >> 8;
  wmod2[((size_t)b * 256 + f) * 2560 + 2304 + c] = (_Float16)w1x1[f * 256 + c];
}

// ---------------------------------------------------------------------------
// bilinear x2 upsample (half-pixel, edge clamp), fp32 -> fp16
// total threads = 8*256*128*128
// ---------------------------------------------------------------------------
__global__ void upsample_kernel(const float* __restrict__ x,
                                _Float16* __restrict__ xup) {
  size_t idx = (size_t)blockIdx.x * blockDim.x + threadIdx.x;
  int j = (int)(idx & 127);
  size_t t = idx >> 7;
  int i = (int)(t & 127);
  size_t bc = t >> 7;
  float ph = i * 0.5f - 0.25f, pw = j * 0.5f - 0.25f;
  int i0 = (int)floorf(ph), j0 = (int)floorf(pw);
  float fh = ph - (float)i0, fw = pw - (float)j0;
  int i0c = i0 < 0 ? 0 : i0;           int i1c = i0 + 1 > 63 ? 63 : i0 + 1;
  int j0c = j0 < 0 ? 0 : j0;           int j1c = j0 + 1 > 63 ? 63 : j0 + 1;
  const float* src = x + bc * 4096;
  float v00 = src[i0c * 64 + j0c], v01 = src[i0c * 64 + j1c];
  float v10 = src[i1c * 64 + j0c], v11 = src[i1c * 64 + j1c];
  float v = (1.f - fh) * ((1.f - fw) * v00 + fw * v01)
          +        fh  * ((1.f - fw) * v10 + fw * v11);
  xup[idx] = (_Float16)v;
}

// ---------------------------------------------------------------------------
// Implicit-GEMM modulated 3x3 conv with WMMA f16->f32.
//   M = 256 (F, 2 tiles of 128), N = 128*128 (one image row per block),
//   K = Ktot (2304, or 2560 with folded 1x1 residual tail).
// Double-buffered LDS ping-pong; A tile streamed with CDNA5 async
// global->LDS (ASYNCcnt) when available; B tile is implicit im2col built
// transposed in LDS. 8 wave32s per block, each owns a 64x32 tile (4x2 WMMA
// accumulators, 8 v_wmma per K-step). Epilogue: +bias (noise), leaky-ReLU.
// grid: (128, 2, 8)  block: (256)
// ---------------------------------------------------------------------------
__global__ void __launch_bounds__(256)
conv_gemm_kernel(const _Float16* __restrict__ Wmod,
                 const _Float16* __restrict__ Xin,
                 const _Float16* __restrict__ Xtail,
                 const float* __restrict__ bias,
                 _Float16* __restrict__ Oh,
                 float* __restrict__ Of,
                 int Ktot, int mode) {
  __shared__ _Float16 At[2][128][32];   // [buf][m][k]
  __shared__ _Float16 Bt[2][128][32];   // [buf][n][k] (transposed)

  const int h     = blockIdx.x;      // output row
  const int mtile = blockIdx.y;      // 0..1 (which 128 of F)
  const int b     = blockIdx.z;      // batch
  const int tid   = threadIdx.x;
  const int lane  = tid & 31, wid = tid >> 5;
  const int wm    = wid & 1,  wn  = wid >> 1;     // 2x4 wave grid
  const int row   = lane & 15, hs = lane >> 4;

  v8f acc[4][2] = {};

  const _Float16* Wb = Wmod + ((size_t)b * 256 + (size_t)mtile * 128) * (size_t)Ktot;
  const int a_row = tid & 127, a_half = tid >> 7;  // A fill: 16 f16 per thread
  const int b_k   = tid & 31,  b_g    = tid >> 5;  // B fill: row k, 16 cols

  auto fill = [&](int buf, int k0) {
    // ---- A tile: async global->LDS (2 x b128 per thread) or sync fallback
    const _Float16* gA = &Wb[(size_t)a_row * Ktot + k0 + a_half * 16];
    _Float16* lA = &At[buf][a_row][a_half * 16];
#if USE_ASYNC
    __builtin_amdgcn_global_load_async_to_lds_b128(
        (GLOBAL_AS v4i*)gA, (LDS_AS v4i*)lA, 0, 0);
    __builtin_amdgcn_global_load_async_to_lds_b128(
        (GLOBAL_AS v4i*)(gA + 8), (LDS_AS v4i*)(lA + 8), 0, 0);
#else
    *(v16h*)lA = *(const v16h*)gA;
#endif

    // ---- B tile: implicit im2col, stored transposed
    int k = k0 + b_k;
    if (k < 2304) {
      int c  = k / 9, r = k - c * 9;
      int kh = r / 3 - 1;
      int kw = (r - (r / 3) * 3) - 1;
      int hh = h + kh;
      bool hok = (hh >= 0) && (hh < 128);
      const _Float16* src = Xin + (((size_t)b * 256 + c) * 128 + hh) * 128;
#pragma unroll
      for (int j = 0; j < 16; ++j) {
        int n  = b_g * 16 + j;
        int ww = n + kw;
        _Float16 v = (_Float16)0.f;
        if (hok && ww >= 0 && ww < 128) v = src[ww];
        Bt[buf][n][b_k] = v;
      }
    } else {                                   // folded 1x1 residual tail
      int c2 = k - 2304;
      const _Float16* src = Xtail + (((size_t)b * 256 + c2) * 128 + h) * 128;
#pragma unroll
      for (int j = 0; j < 16; ++j) {
        int n = b_g * 16 + j;
        Bt[buf][n][b_k] = src[n];
      }
    }
  };

  auto compute = [&](int buf) {
    v16h af[4], bf[2];
#pragma unroll
    for (int i = 0; i < 4; ++i)
      af[i] = *(const v16h*)&At[buf][wm * 64 + i * 16 + row][hs * 16];
#pragma unroll
    for (int j = 0; j < 2; ++j)
      bf[j] = *(const v16h*)&Bt[buf][wn * 32 + j * 16 + row][hs * 16];
#pragma unroll
    for (int i = 0; i < 4; ++i)
#pragma unroll
      for (int j = 0; j < 2; ++j)
        acc[i][j] = __builtin_amdgcn_wmma_f32_16x16x32_f16(
            false, af[i], false, bf[j], (short)0, acc[i][j], false, false);
  };

  const int nsteps = Ktot >> 5;
  fill(0, 0);
#if USE_ASYNC
  __builtin_amdgcn_s_wait_asynccnt(0);
#endif
  __syncthreads();

  int buf = 0;
  for (int s = 0; s < nsteps; ++s) {
    if (s + 1 < nsteps) fill(buf ^ 1, (s + 1) << 5);   // prefetch next tile
    compute(buf);                                       // overlap with fills
#if USE_ASYNC
    __builtin_amdgcn_s_wait_asynccnt(0);
#endif
    __syncthreads();
    buf ^= 1;
  }

  // ---- epilogue: + noise bias, leaky ReLU
#pragma unroll
  for (int i = 0; i < 4; ++i) {
#pragma unroll
    for (int j = 0; j < 2; ++j) {
#pragma unroll
      for (int r = 0; r < 8; ++r) {
        int f = mtile * 128 + wm * 64 + i * 16 + hs * 8 + r;
        int w = wn * 32 + j * 16 + row;
        float v = acc[i][j][r] + bias[f];
        v = (v >= 0.f) ? v : LRELU_SLOPE * v;
        size_t o = (((size_t)b * 256 + f) * 128 + h) * 128 + w;
        if (mode == 0) Oh[o] = (_Float16)v;
        else           Of[o] = v;
      }
    }
  }
}

// ---------------------------------------------------------------------------
extern "C" void kernel_launch(void* const* d_in, const int* in_sizes, int n_in,
                              void* d_out, int out_size, void* d_ws, size_t ws_size,
                              hipStream_t stream) {
  (void)in_sizes; (void)n_in; (void)out_size; (void)ws_size;

  const float* x         = (const float*)d_in[0];
  const float* istyle    = (const float*)d_in[1];
  const float* w_style1  = (const float*)d_in[2];
  const float* b_style1  = (const float*)d_in[3];
  /* w_noise1 = d_in[4] unused: inoise == 0 */
  const float* b_noise1  = (const float*)d_in[5];
  const float* w_conv1   = (const float*)d_in[6];
  const float* w_style2  = (const float*)d_in[7];
  const float* b_style2  = (const float*)d_in[8];
  /* w_noise2 = d_in[9] unused */
  const float* b_noise2  = (const float*)d_in[10];
  const float* w_conv2   = (const float*)d_in[11];
  const float* w_conv1x1 = (const float*)d_in[12];
  float* out = (float*)d_out;

  char* ws = (char*)d_ws;
  float*    s1p   = (float*)   (ws + 0);          //  8 KB
  float*    s2p   = (float*)   (ws + 8192);       //  8 KB
  float*    d1    = (float*)   (ws + 16384);      //  8 KB
  float*    d2    = (float*)   (ws + 24576);      //  8 KB
  float*    wsq1  = (float*)   (ws + 32768);      // 256 KB
  float*    wsq2  = (float*)   (ws + 294912);     // 256 KB
  _Float16* wmod1 = (_Float16*)(ws + 557056);     // 8*256*2304 f16 = 9.44 MB
  _Float16* wmod2 = (_Float16*)(ws + 9994240);    // 8*256*2560 f16 = 10.49 MB
  _Float16* xup   = (_Float16*)(ws + 20480000);   // 8*256*128*128 f16 = 67.1 MB
  _Float16* h1    = (_Float16*)(ws + 87588864);   // 67.1 MB
  // total workspace: ~154.7 MB

  // styles (+1 baked in)
  style_kernel<<<8, 256, 0, stream>>>(istyle, w_style1, b_style1, s1p, 512);
  style_kernel<<<8, 256, 0, stream>>>(istyle, w_style2, b_style2, s2p, 512);
  // per-(f,c) sum of squared 3x3 weights
  wsq_kernel<<<256, 256, 0, stream>>>(w_conv1, wsq1);
  wsq_kernel<<<256, 256, 0, stream>>>(w_conv2, wsq2);
  // demod coefficients
  dcoef_kernel<<<8, 256, 0, stream>>>(s1p, wsq1, d1);
  dcoef_kernel<<<8, 256, 0, stream>>>(s2p, wsq2, d2);
  // fp16 modulated+demodulated weights
  modw_kernel<<<18432, 256, 0, stream>>>(w_conv1, s1p, d1, wmod1, 2304);
  modw_kernel<<<18432, 256, 0, stream>>>(w_conv2, s2p, d2, wmod2, 2560);
  tail_kernel<<<2048, 256, 0, stream>>>(w_conv1x1, wmod2);
  // bilinear x2 upsample to fp16
  upsample_kernel<<<131072, 256, 0, stream>>>(x, xup);

  dim3 grid(128, 2, 8);
  // conv1: h1 = lrelu(conv(xup, wmod1) + b_noise1)   (fp16 out)
  conv_gemm_kernel<<<grid, 256, 0, stream>>>(wmod1, xup, xup, b_noise1,
                                             h1, nullptr, 2304, 0);
  // conv2 (+folded 1x1 residual): out = lrelu(conv(h1,wmod2)+res+b_noise2)
  conv_gemm_kernel<<<grid, 256, 0, stream>>>(wmod2, h1, xup, b_noise2,
                                             nullptr, out, 2560, 1);
}
